// MSTM_55250459295992
// MI455X (gfx1250) — matrix operand order repaired and checked by reading, not compiled, not run on hardware
//
#include <hip/hip_runtime.h>
#include <hip/hip_bf16.h>

// Problem constants (from reference): B=8, C=256, H=W=128 -> HW=16384, HEADS=4, HEAD_DIM=64
#define HW   16384
#define CH   256
#define NB   8
#define NH   4
#define HD   64
#define OC3  768   // 3*C rows of Wqkv

typedef __bf16 bf16;
typedef __attribute__((ext_vector_type(4)))  float  f32x4;
typedef __attribute__((ext_vector_type(4)))  __bf16 bf16x4;
typedef __attribute__((ext_vector_type(8)))  __bf16 bf16x8;
typedef __attribute__((ext_vector_type(16))) __bf16 v16bf;
typedef __attribute__((ext_vector_type(8)))  float  v8f;
typedef __attribute__((ext_vector_type(4)))  unsigned u32x4;
typedef __attribute__((ext_vector_type(8)))  int      i32x8;
typedef __attribute__((ext_vector_type(4)))  int      i32x4;

#define AS1 __attribute__((address_space(1)))
#define AS3 __attribute__((address_space(3)))

// CDNA5 async/tensor data movers (compile-probed; guarded so absence falls back)
#if defined(__gfx1250__) && __has_builtin(__builtin_amdgcn_tensor_load_to_lds) && \
    __has_builtin(__builtin_amdgcn_s_wait_tensorcnt)
#define HAVE_TDM 1
#endif
#if defined(__gfx1250__) && __has_builtin(__builtin_amdgcn_global_load_async_to_lds_b128)
#define HAVE_ASYNC 1
#endif

__device__ __forceinline__ void wait_asynccnt0() {
#if defined(__gfx1250__)
#if __has_builtin(__builtin_amdgcn_s_wait_asynccnt)
  __builtin_amdgcn_s_wait_asynccnt(0);
#else
  asm volatile("s_wait_asynccnt 0" ::: "memory");
#endif
#endif
}

__device__ __forceinline__ v8f wmma_bf16(v16bf a, v16bf b, v8f c) {
  // D = A(16x32 bf16) x B(32x16 bf16) + C(16x16 f32)
  return __builtin_amdgcn_wmma_f32_16x16x32_bf16(
      /*neg_a=*/false, a, /*neg_b=*/false, b,
      /*c_mod=*/(short)0, c, /*reuse_a=*/false, /*reuse_b=*/false);
}

__device__ __forceinline__ v16bf pack2(bf16x8 l, bf16x8 h) {
  return __builtin_shufflevector(l, h, 0,1,2,3,4,5,6,7,8,9,10,11,12,13,14,15);
}

// ---------------------------------------------------------------- fp32 -> bf16
__global__ void k_cvt(const float* __restrict__ in, bf16* __restrict__ out, long n4) {
  long i = (long)blockIdx.x * blockDim.x + threadIdx.x;
  long stride = (long)gridDim.x * blockDim.x;
  for (; i < n4; i += stride) {
    f32x4 v = ((const f32x4*)in)[i];
    bf16x4 o;
    o.x = (bf16)v.x; o.y = (bf16)v.y; o.z = (bf16)v.z; o.w = (bf16)v.w;
    ((bf16x4*)out)[i] = o;
  }
}

__global__ void k_zero(float* __restrict__ p, int n) {
  int i = blockIdx.x * blockDim.x + threadIdx.x;
  if (i < n) p[i] = 0.0f;
}

// ---------------------------------------------------------------- GEMM: Y[b] = W(MxK) @ X[b](KxHW) + bias
// Workgroup (8 waves) = 64(M) x 128(N) tile; per K-step of 32 the block stages a
// 32x128 bf16 tile of X into LDS (TDM descriptor-DMA if available, else async
// load-to-LDS, else coalesced b128 copies), then each wave builds its B operands
// from LDS and runs 2x2 WMMAs with f32 accumulation.
template <typename TB, typename TO>
__global__ void k_gemm(const bf16* __restrict__ W, const TB* __restrict__ X,
                       const float* __restrict__ bias, TO* __restrict__ Y,
                       int M, int K) {
  __shared__ bf16 tx[32][128 + 8];  // +8 bf16 pad: row stride 272B -> bank-shift 4

  const int tid  = threadIdx.x;
  const int lane = tid & 31, wave = tid >> 5;
  const int lo = lane & 15, hi = lane >> 4;
  const int wm = wave >> 2, wn = wave & 3;
  const int b    = blockIdx.z;
  const int m0   = blockIdx.x * 64 + wm * 32;
  const int nblk = blockIdx.y * 128;
  const TB* Xb = X + (size_t)b * K * HW;

  // staging role (fallback paths): each thread copies 16 elements
  const int sk = tid >> 3, sseg = tid & 7;

  v8f acc[2][2] = {};

  for (int k0 = 0; k0 < K; k0 += 32) {
    // ---- stage X tile (32 x 128) -> LDS
    if constexpr (sizeof(TB) == 2) {
#if defined(HAVE_TDM)
      if (wave == 0) {
        // Tensor DMA descriptor (ISA 8.3-8.6): 2D tensor, 2B elems,
        // tile 128(dim0=n) x 32(dim1=c), row stride HW, LDS pad 16B per 256B.
        unsigned lds_base = (unsigned)(__UINTPTR_TYPE__)(AS3 void*)&tx[0][0];
        unsigned long long ga =
            (unsigned long long)(__UINTPTR_TYPE__)(Xb + (size_t)k0 * HW + nblk);
        u32x4 g0;
        g0.x = 1u;                                   // count=1 (valid)
        g0.y = lds_base;                             // lds_addr
        g0.z = (unsigned)ga;                         // global_addr[31:0]
        g0.w = (unsigned)((ga >> 32) & 0x1FFFFFFu) | (2u << 30);  // addr[56:32] | type=2
        i32x8 g1;
        g1[0] = (1 << 16)      // data_size = 2B
              | (1 << 20)      // pad_enable
              | (5 << 22)      // pad_interval: 64 DWORDs = 256B
              | (3 << 25);     // pad_amount: 4 DWORDs = 16B
        g1[1] = (int)(16384u << 16);          // tensor_dim0[15:0]=HW
        g1[2] = (int)((unsigned)K << 16);     // dim0[31:16]=0 | tensor_dim1[15:0]=K
        g1[3] = (int)(128u << 16);            // dim1[31:16]=0 | tile_dim0=128
        g1[4] = 32;                           // tile_dim1=32, tile_dim2=0
        g1[5] = 16384;                        // tensor_dim0_stride lo32 = HW
        g1[6] = (int)(16384u << 16);          // stride0 hi=0 | tensor_dim1_stride lo16
        g1[7] = 0;
        i32x4 gz = {0, 0, 0, 0};
        i32x8 gz8 = {0, 0, 0, 0, 0, 0, 0, 0};
        __builtin_amdgcn_tensor_load_to_lds(g0, g1, gz, gz, gz8, 0);
        __builtin_amdgcn_s_wait_tensorcnt(0);
      }
#elif defined(HAVE_ASYNC)
      {
        const TB* src = Xb + (size_t)(k0 + sk) * HW + nblk + sseg * 16;
        bf16* dst = &tx[sk][sseg * 16];
        __builtin_amdgcn_global_load_async_to_lds_b128(
            (AS1 i32x4*)src, (AS3 i32x4*)dst, 0, 0);
        __builtin_amdgcn_global_load_async_to_lds_b128(
            (AS1 i32x4*)(src + 8), (AS3 i32x4*)(dst + 8), 0, 0);
        wait_asynccnt0();
      }
#else
      {
        const TB* src = Xb + (size_t)(k0 + sk) * HW + nblk + sseg * 16;
        bf16* dst = &tx[sk][sseg * 16];
        *(bf16x8*)dst       = *(const bf16x8*)(const void*)src;
        *(bf16x8*)(dst + 8) = *(const bf16x8*)(const void*)(src + 8);
      }
#endif
    } else {
      const TB* src = Xb + (size_t)(k0 + sk) * HW + nblk + sseg * 16;
      bf16* dst = &tx[sk][sseg * 16];
#pragma unroll
      for (int j = 0; j < 16; j += 4) {
        f32x4 v = *(const f32x4*)(const void*)(src + j);
        dst[j + 0] = (bf16)v.x; dst[j + 1] = (bf16)v.y;
        dst[j + 2] = (bf16)v.z; dst[j + 3] = (bf16)v.w;
      }
    }
    __syncthreads();

    // A operand: 16x32 bf16 from W; elem j -> K = k0 + (j/8)*16 + hi*8 + j%8
    v16bf a[2];
#pragma unroll
    for (int ti = 0; ti < 2; ++ti) {
      const bf16* p = W + (size_t)(m0 + ti * 16 + lo) * K + k0 + hi * 8;
      a[ti] = pack2(*(const bf16x8*)p, *(const bf16x8*)(p + 16));
    }
    // B operand from LDS: col N = lane%16; elem j -> K-row = (j/8)*16 + hi*8 + j%8
    v16bf bm[2];
#pragma unroll
    for (int tj = 0; tj < 2; ++tj) {
      v16bf t;
#pragma unroll
      for (int e = 0; e < 16; ++e) t[e] = tx[(e >> 3) * 16 + hi * 8 + (e & 7)][wn * 32 + tj * 16 + lo];
      bm[tj] = t;
    }
#pragma unroll
    for (int ti = 0; ti < 2; ++ti)
#pragma unroll
      for (int tj = 0; tj < 2; ++tj)
        acc[ti][tj] = wmma_bf16(a[ti], bm[tj], acc[ti][tj]);

    __syncthreads();
  }

  // D layout: VGPR r -> M = r + (hi?8:0), N = lane%16
#pragma unroll
  for (int ti = 0; ti < 2; ++ti)
#pragma unroll
    for (int tj = 0; tj < 2; ++tj)
#pragma unroll
      for (int r = 0; r < 8; ++r) {
        int row = m0 + ti * 16 + r + hi * 8;
        int col = nblk + wn * 32 + tj * 16 + lo;
        float v = acc[ti][tj][r] + bias[row];
        Y[(size_t)b * M * HW + (size_t)row * HW + col] = (TO)v;
      }
}

// ---------------------------------------------------------------- S = q @ k^T  (per b,h: 64x64, K=16384)
// grid (NH, NB, 8 K-splits); wave: strip = d-rows, half = K sub-split. Partials atomicAdd'ed.
__global__ void k_qk(const bf16* __restrict__ qkv, float* __restrict__ Sg) {
  const int h = blockIdx.x, b = blockIdx.y, split = blockIdx.z;
  const int lane = threadIdx.x & 31, wave = threadIdx.x >> 5;
  const int lo = lane & 15, hi = lane >> 4;
  const int strip = wave & 3, half = wave >> 2;

  const bf16* qb = qkv + ((size_t)b * OC3 + h * HD) * HW;
  const bf16* kb = qkv + ((size_t)b * OC3 + CH + h * HD) * HW;
  const int nbase = split * 2048 + half * 1024;

  v8f acc[4] = {};
  for (int n0 = nbase; n0 < nbase + 1024; n0 += 32) {
    const bf16* pq = qb + (size_t)(strip * 16 + lo) * HW + n0 + hi * 8;
    v16bf a = pack2(*(const bf16x8*)pq, *(const bf16x8*)(pq + 16));
#pragma unroll
    for (int t = 0; t < 4; ++t) {
      // B = k^T: col e = t*16+lo, K run contiguous along n
      const bf16* pk = kb + (size_t)(t * 16 + lo) * HW + n0 + hi * 8;
      v16bf bm = pack2(*(const bf16x8*)pk, *(const bf16x8*)(pk + 16));
      acc[t] = wmma_bf16(a, bm, acc[t]);
    }
  }

  float* Sbh = Sg + (size_t)(b * NH + h) * HD * HD;
#pragma unroll
  for (int t = 0; t < 4; ++t)
#pragma unroll
    for (int r = 0; r < 8; ++r) {
      int row = strip * 16 + r + hi * 8;
      int col = t * 16 + lo;
      atomicAdd(&Sbh[row * HD + col], acc[t][r]);
    }
}

// ---------------------------------------------------------------- row softmax (scale, max-subtract) -> bf16 P
__global__ void k_softmax(const float* __restrict__ Sg, bf16* __restrict__ Pg) {
  const int bh = blockIdx.x;     // NB*NH blocks
  const int row = threadIdx.x;   // 64 threads = rows
  const float* s = Sg + (size_t)bh * HD * HD + row * HD;
  bf16* p = Pg + (size_t)bh * HD * HD + row * HD;
  const float scale = 0.125f;    // HEAD_DIM^-0.5 = 1/8

  float vals[HD];
  float m = -3.4e38f;
#pragma unroll
  for (int e = 0; e < HD; ++e) { float v = s[e] * scale; vals[e] = v; m = fmaxf(m, v); }
  float sum = 0.0f;
#pragma unroll
  for (int e = 0; e < HD; ++e) { float v = __expf(vals[e] - m); vals[e] = v; sum += v; }
  float inv = 1.0f / sum;
#pragma unroll
  for (int e = 0; e < HD; ++e) p[e] = (bf16)(vals[e] * inv);
}

// ---------------------------------------------------------------- ctx = P @ v  (64x16384 per b,h, K=64)
// Block stages a 64(e) x 32(n) tile of v into LDS (async load-to-LDS when
// available), each wave (strip = d-rows, nh = n-half) computes one 16x16 tile.
__global__ void k_pv(const bf16* __restrict__ qkv, const bf16* __restrict__ Pg,
                     float* __restrict__ ctx) {
  __shared__ bf16 tv[64][32 + 8];  // pad -> row stride 80B, bank-shift 20

  const int tid = threadIdx.x;
  const int h = blockIdx.x, b = blockIdx.y, split = blockIdx.z;
  const int lane = tid & 31, wave = tid >> 5;
  const int lo = lane & 15, hi = lane >> 4;
  const int strip = wave & 3, nh = wave >> 2;

  const bf16* vb = qkv + ((size_t)b * OC3 + 2 * CH + h * HD) * HW;
  const bf16* P  = Pg + (size_t)(b * NH + h) * HD * HD;

  // A operands (P, M=d rows, K=e): loaded once per wave
  v16bf aP[2];
#pragma unroll
  for (int kk = 0; kk < 2; ++kk) {
    const bf16* pp = P + (size_t)(strip * 16 + lo) * HD + kk * 32 + hi * 8;
    aP[kk] = pack2(*(const bf16x8*)pp, *(const bf16x8*)(pp + 16));
  }

  float* cb = ctx + ((size_t)b * CH + h * HD) * HW;
  const int nbase = split * 2048;

  // staging role: row e = tid>>2, segment = tid&3 (8 bf16 = 16B each)
  const int se = tid >> 2, sseg = tid & 3;

  for (int n0 = nbase; n0 < nbase + 2048; n0 += 32) {
#if defined(HAVE_ASYNC)
    __builtin_amdgcn_global_load_async_to_lds_b128(
        (AS1 i32x4*)(vb + (size_t)se * HW + n0 + sseg * 8),
        (AS3 i32x4*)&tv[se][sseg * 8], 0, 0);
    wait_asynccnt0();
#else
    *(bf16x8*)&tv[se][sseg * 8] =
        *(const bf16x8*)(vb + (size_t)se * HW + n0 + sseg * 8);
#endif
    __syncthreads();

    v8f acc = {};
#pragma unroll
    for (int kk = 0; kk < 2; ++kk) {
      v16bf bm;
#pragma unroll
      for (int e = 0; e < 16; ++e) bm[e] = tv[kk * 32 + (e >> 3) * 16 + hi * 8 + (e & 7)][nh * 16 + lo];
      acc = wmma_bf16(aP[kk], bm, acc);
    }
#pragma unroll
    for (int r = 0; r < 8; ++r) {
      int d = strip * 16 + r + hi * 8;
      cb[(size_t)d * HW + n0 + nh * 16 + lo] = acc[r];
    }
    __syncthreads();
  }
}

// ----------------------------------------------------------------
extern "C" void kernel_launch(void* const* d_in, const int* in_sizes, int n_in,
                              void* d_out, int out_size, void* d_ws, size_t ws_size,
                              hipStream_t stream) {
  const float* x     = (const float*)d_in[0];  // (B, C, H, W)
  const float* Wqkv  = (const float*)d_in[1];  // (3C, C)
  const float* bqkv  = (const float*)d_in[2];  // (3C)
  const float* Wproj = (const float*)d_in[3];  // (C, C)
  const float* bproj = (const float*)d_in[4];  // (C)
  float* out = (float*)d_out;

  char* ws = (char*)d_ws;
  size_t off = 0;
  auto alloc = [&](size_t bytes) -> void* {
    void* p = ws + off;
    off += (bytes + 255) & ~(size_t)255;
    return p;
  };
  bf16*  xb   = (bf16*) alloc((size_t)NB * CH * HW * sizeof(bf16));
  bf16*  Wqb  = (bf16*) alloc((size_t)OC3 * CH * sizeof(bf16));
  bf16*  Wpb  = (bf16*) alloc((size_t)CH * CH * sizeof(bf16));
  bf16*  qkvb = (bf16*) alloc((size_t)NB * OC3 * HW * sizeof(bf16));
  float* Sg   = (float*)alloc((size_t)NB * NH * HD * HD * sizeof(float));
  bf16*  Pg   = (bf16*) alloc((size_t)NB * NH * HD * HD * sizeof(bf16));
  float* ctx  = (float*)alloc((size_t)NB * CH * HW * sizeof(float));

  // Stage bf16 copies of x and the weight matrices.
  k_cvt<<<2048, 256, 0, stream>>>(x, xb, (long)NB * CH * HW / 4);
  k_cvt<<<64, 256, 0, stream>>>(Wqkv, Wqb, (long)OC3 * CH / 4);
  k_cvt<<<64, 256, 0, stream>>>(Wproj, Wpb, (long)CH * CH / 4);
  k_zero<<<(NB * NH * HD * HD + 255) / 256, 256, 0, stream>>>(Sg, NB * NH * HD * HD);

  // qkv = Wqkv @ x + bqkv   (bf16 out, f32 accumulate)
  k_gemm<bf16, bf16><<<dim3(OC3 / 64, HW / 128, NB), 256, 0, stream>>>(
      Wqb, xb, bqkv, qkvb, OC3, CH);

  // S = q @ k^T (split-K atomics), softmax -> P (bf16)
  k_qk<<<dim3(NH, NB, 8), 256, 0, stream>>>(qkvb, Sg);
  k_softmax<<<dim3(NB * NH), HD, 0, stream>>>(Sg, Pg);

  // ctx = P @ v (f32)
  k_pv<<<dim3(NH, NB, 8), 256, 0, stream>>>(qkvb, Pg, ctx);

  // out = Wproj @ ctx + bproj (f32 output)
  k_gemm<float, float><<<dim3(CH / 64, HW / 128, NB), 256, 0, stream>>>(
      Wpb, ctx, bproj, out, CH, CH);
}
